// Classifier_14362370638536
// MI455X (gfx1250) — matrix operand order repaired and checked by reading, not compiled, not run on hardware
//
#include <hip/hip_runtime.h>
#include <hip/hip_bf16.h>

// ---------------------------------------------------------------------------
// GAT (4 layers) + mean pool + MLP head for MI455X (gfx1250, wave32, WMMA).
// Dense matmuls: v_wmma_f32_16x16x32_f16 with f16 operands pre-staged in
// fragment-friendly layouts (A row-major padded, W transposed) so each
// fragment is two unconditional global_load_b128 per lane. Edge softmax and
// aggregation use L2-resident atomics (all node state fits in 192MB L2).
// ---------------------------------------------------------------------------

#define N_NODES  100000
#define N_EDGES  3200000
#define E_TOT    (N_EDGES + N_NODES)   // edges + self loops
#define N_GRAPHS 1024
#define X_DIM    79
#define H_DIM    64
#define NEG_SLOPE 0.2f

typedef __attribute__((ext_vector_type(16))) _Float16 v16h;
typedef __attribute__((ext_vector_type(8)))  _Float16 v8h;
typedef __attribute__((ext_vector_type(8)))  float    v8f;

// ---- helpers ---------------------------------------------------------------

__device__ __forceinline__ unsigned f2ord(float f) {
    unsigned u = __float_as_uint(f);
    return (u & 0x80000000u) ? ~u : (u | 0x80000000u);
}
__device__ __forceinline__ float ord2f(unsigned u) {
    unsigned b = (u & 0x80000000u) ? (u & 0x7FFFFFFFu) : ~u;
    return __uint_as_float(b);
}
__device__ __forceinline__ void edge_sd(const int* __restrict__ ei, long long i,
                                        int& s, int& d) {
    if (i < (long long)N_EDGES) { s = ei[i]; d = ei[(long long)N_EDGES + i]; }
    else { s = d = (int)(i - N_EDGES); }
}

// ---- weight conversion: f32 -> f16, TRANSPOSED (col-major), W0 K pad 79->96

__global__ void k_conv_weights(const float* __restrict__ W0,
                               const float* __restrict__ Wc,
                               _Float16* __restrict__ Wt0,   // 64 cols x 96 k
                               _Float16* __restrict__ Wtc) { // 3 x (64 cols x 64 k)
    int i = blockIdx.x * blockDim.x + threadIdx.x;
    if (i < 64 * 96) {
        int col = i / 96, k = i % 96;
        Wt0[i] = (k < X_DIM) ? (_Float16)W0[k * 64 + col] : (_Float16)0.f;
    }
    int j = i - 64 * 96;
    if (j >= 0 && j < 3 * 64 * 64) {
        int l = j / 4096, r = j % 4096;
        int col = r / 64, k = r % 64;
        Wtc[j] = (_Float16)Wc[l * 4096 + k * 64 + col];
    }
}

// ---- convert x (N x 79 f32) -> Ah (N x 96 f16, zero padded) ----------------

__global__ void k_cvt_x(const float* __restrict__ x, _Float16* __restrict__ Ah) {
    long long i = (long long)blockIdx.x * blockDim.x + threadIdx.x;
    if (i >= (long long)N_NODES * 96) return;
    int row = (int)(i / 96), c = (int)(i % 96);
    Ah[i] = (c < X_DIM) ? (_Float16)x[(size_t)row * X_DIM + c] : (_Float16)0.f;
}

// ---- WMMA GEMM: H[M x 64](f32) = Ah[M x Kp](f16) * Wt(f16, col-major) ------
// Block = 128 threads (4 waves) covering 32 rows x 64 cols. Each wave owns a
// 16-col stripe and computes two stacked 16x16 tiles (shared B fragment).
// Fragment element mapping per CDNA5 ISA 7.12.2; all loads are b128, no
// branches, EXEC all-ones throughout (grid is exact: 100000/32 = 3125).

__global__ void k_gemm_wmma(const _Float16* __restrict__ Ah,
                            const _Float16* __restrict__ Wt,
                            float* __restrict__ H, int Kp) {
    const int blk  = blockIdx.x;
    const int wave = threadIdx.x >> 5;
    const int lane = threadIdx.x & 31;
    const int half = lane >> 4;
    const int l15  = lane & 15;
    const int col  = wave * 16 + l15;

    const _Float16* a0p = Ah + ((size_t)blk * 32 + l15) * Kp;  // rows 0..15
    const _Float16* a1p = a0p + (size_t)16 * Kp;               // rows 16..31
    const _Float16* bp  = Wt + (size_t)col * Kp;

    v8f acc0 = {}, acc1 = {};
    for (int kt = 0; kt < Kp; kt += 32) {
        // B (32x16): e<8 -> k = kt + e + 8*half ; e>=8 -> k = kt+16 + (e-8) + 8*half
        v16h b;
        {
            v8h b0 = *(const v8h*)(bp + kt + (half << 3));
            v8h b1 = *(const v8h*)(bp + kt + 16 + (half << 3));
#pragma unroll
            for (int e = 0; e < 8; ++e) { b[e] = b0[e]; b[e + 8] = b1[e]; }
        }
        // A (16x32): e<8 -> k = kt + e + 8*half ; e>=8 -> k = kt+16 + (e-8) + 8*half
        v16h a;
        {
            v8h x0 = *(const v8h*)(a0p + kt + (half << 3));
            v8h x1 = *(const v8h*)(a0p + kt + 16 + (half << 3));
#pragma unroll
            for (int e = 0; e < 8; ++e) { a[e] = x0[e]; a[e + 8] = x1[e]; }
        }
        acc0 = __builtin_amdgcn_wmma_f32_16x16x32_f16(
            false, a, false, b, (short)0, acc0, false, false);
        {
            v8h x0 = *(const v8h*)(a1p + kt + (half << 3));
            v8h x1 = *(const v8h*)(a1p + kt + 16 + (half << 3));
#pragma unroll
            for (int e = 0; e < 8; ++e) { a[e] = x0[e]; a[e + 8] = x1[e]; }
        }
        acc1 = __builtin_amdgcn_wmma_f32_16x16x32_f16(
            false, a, false, b, (short)0, acc1, false, false);
    }
    // C/D: vgpr v -> M = v + 8*half, N = lane&15
    const size_t r0 = (size_t)blk * 32 + (half << 3);
#pragma unroll
    for (int v = 0; v < 8; ++v) {
        H[(r0 + v) * 64 + col]      = acc0[v];
        H[(r0 + 16 + v) * 64 + col] = acc1[v];
    }
}

// ---- per-node attention dots: asn = h.a_s, adn = h.a_d ---------------------

__global__ void k_node_dots(const float* __restrict__ h,
                            const float* __restrict__ as,
                            const float* __restrict__ ad,
                            float* __restrict__ asn, float* __restrict__ adn) {
    int i = blockIdx.x * blockDim.x + threadIdx.x;
    if (i >= N_NODES) return;
    const float* hr = h + (size_t)i * 64;
    float s = 0.f, d = 0.f;
#pragma unroll
    for (int k = 0; k < 64; ++k) { float v = hr[k]; s += v * as[k]; d += v * ad[k]; }
    asn[i] = s; adn[i] = d;
}

// ---- init per-layer scratch ------------------------------------------------

__global__ void k_zero_layer(float* __restrict__ out,      // N x 64 accum
                             unsigned* __restrict__ emax,  // N
                             float* __restrict__ denom) {  // N
    int i = blockIdx.x * blockDim.x + threadIdx.x;
    if (i < N_NODES * 64) out[i] = 0.f;
    if (i < N_NODES) { emax[i] = 0u; denom[i] = 0.f; }
}

// ---- edge pass 1: score + segment max --------------------------------------

__global__ void k_edge_score(const int* __restrict__ ei,
                             const float* __restrict__ asn,
                             const float* __restrict__ adn,
                             float* __restrict__ p,
                             unsigned* __restrict__ emax) {
    long long i = (long long)blockIdx.x * blockDim.x + threadIdx.x;
    if (i >= (long long)E_TOT) return;
    int s, d; edge_sd(ei, i, s, d);
    float e = asn[s] + adn[d];
    e = (e > 0.f) ? e : NEG_SLOPE * e;       // leaky_relu
    p[i] = e;
    atomicMax(&emax[d], f2ord(e));
}

// ---- edge pass 2: exp + segment sum ----------------------------------------

__global__ void k_edge_exp(const int* __restrict__ ei,
                           float* __restrict__ p,
                           const unsigned* __restrict__ emax,
                           float* __restrict__ denom) {
    long long i = (long long)blockIdx.x * blockDim.x + threadIdx.x;
    if (i >= (long long)E_TOT) return;
    int s, d; edge_sd(ei, i, s, d);
    float v = __expf(p[i] - ord2f(emax[d]));
    p[i] = v;
    atomicAdd(&denom[d], v);
}

// ---- edge pass 3: one wave per edge, 64-wide weighted scatter-add ----------

__global__ void k_edge_aggr(const int* __restrict__ ei,
                            const float* __restrict__ p,
                            const float* __restrict__ denom,
                            const float* __restrict__ h,
                            float* __restrict__ out) {
    long long t = (long long)blockIdx.x * blockDim.x + threadIdx.x;
    long long e = t >> 5;
    if (e >= (long long)E_TOT) return;        // uniform across the wave
    int lane = threadIdx.x & 31;
    int s, d; edge_sd(ei, e, s, d);
    float alpha = p[e] / denom[d];
    float2 v = ((const float2*)(h + (size_t)s * 64))[lane];
    atomicAdd(&out[(size_t)d * 64 + lane * 2 + 0], alpha * v.x);
    atomicAdd(&out[(size_t)d * 64 + lane * 2 + 1], alpha * v.y);
}

// ---- finalize: h = relu(out + bias), also emit f16 copy for next GEMM ------

__global__ void k_finalize(float* __restrict__ h, const float* __restrict__ bias,
                           _Float16* __restrict__ Ah) {
    int i = blockIdx.x * blockDim.x + threadIdx.x;
    if (i >= N_NODES * 64) return;
    float v = h[i] + bias[i & 63];
    v = v > 0.f ? v : 0.f;
    h[i]  = v;
    Ah[i] = (_Float16)v;   // row stride 64 for layers >= 1
}

// ---- pooling ---------------------------------------------------------------

__global__ void k_zero_pool(float* __restrict__ gsum, float* __restrict__ gcnt) {
    int i = blockIdx.x * blockDim.x + threadIdx.x;
    if (i < N_GRAPHS * 64) gsum[i] = 0.f;
    if (i < N_GRAPHS) gcnt[i] = 0.f;
}

__global__ void k_pool_acc(const float* __restrict__ h, const int* __restrict__ batch,
                           float* __restrict__ gsum, float* __restrict__ gcnt) {
    long long i = (long long)blockIdx.x * blockDim.x + threadIdx.x;
    if (i >= (long long)N_NODES * 64) return;
    int node = (int)(i >> 6), c = (int)(i & 63);
    int g = batch[node];
    atomicAdd(&gsum[(size_t)g * 64 + c], h[i]);
    if (c == 0) atomicAdd(&gcnt[g], 1.0f);
}

__global__ void k_pool_div(float* __restrict__ gsum, const float* __restrict__ gcnt) {
    int i = blockIdx.x * blockDim.x + threadIdx.x;
    if (i >= N_GRAPHS * 64) return;
    float c = gcnt[i >> 6];
    gsum[i] /= (c > 1.f ? c : 1.f);
}

// ---- MLP head (tiny: 1024 rows) --------------------------------------------

__global__ void k_mlp(const float* __restrict__ in, const float* __restrict__ w,
                      const float* __restrict__ b, float* __restrict__ out,
                      int Kin, int Ko) {
    int i = blockIdx.x * blockDim.x + threadIdx.x;
    if (i >= N_GRAPHS * Ko) return;
    int g = i / Ko, j = i % Ko;
    const float* r = in + (size_t)g * Kin;
    float acc = b[j];
    for (int k = 0; k < Kin; ++k) acc += r[k] * w[k * Ko + j];
    out[i] = acc;
}

__global__ void k_mlp3_sigmoid(const float* __restrict__ in, const float* __restrict__ w,
                               const float* __restrict__ b, float* __restrict__ out) {
    int g = blockIdx.x * blockDim.x + threadIdx.x;
    if (g >= N_GRAPHS) return;
    const float* r = in + (size_t)g * 32;
    float acc = b[0];
#pragma unroll
    for (int k = 0; k < 32; ++k) acc += r[k] * w[k];
    out[g] = 1.f / (1.f + __expf(-acc));
}

// ---------------------------------------------------------------------------

extern "C" void kernel_launch(void* const* d_in, const int* in_sizes, int n_in,
                              void* d_out, int out_size, void* d_ws, size_t ws_size,
                              hipStream_t stream) {
    (void)in_sizes; (void)n_in; (void)out_size; (void)ws_size;

    const float* x   = (const float*)d_in[0];
    const int*   ei  = (const int*)  d_in[1];
    const int*   bat = (const int*)  d_in[2];
    const float* W0  = (const float*)d_in[3];
    const float* a0s = (const float*)d_in[4];
    const float* a0d = (const float*)d_in[5];
    const float* b0  = (const float*)d_in[6];
    const float* Wc  = (const float*)d_in[7];
    const float* acs = (const float*)d_in[8];
    const float* acd = (const float*)d_in[9];
    const float* bc  = (const float*)d_in[10];
    const float* l1w = (const float*)d_in[11];
    const float* l1b = (const float*)d_in[12];
    const float* l2w = (const float*)d_in[13];
    const float* l2b = (const float*)d_in[14];
    const float* l3w = (const float*)d_in[15];
    const float* l3b = (const float*)d_in[16];

    // ---- workspace layout (256B aligned) ----
    char* ws = (char*)d_ws;
    size_t off = 0;
    auto carve = [&](size_t bytes) { void* pp = ws + off; off = (off + bytes + 255) & ~(size_t)255; return pp; };
    float*    hA    = (float*)   carve((size_t)N_NODES * 64 * 4);  // gemm out / edge src feats
    float*    hB    = (float*)   carve((size_t)N_NODES * 64 * 4);  // aggregation accum / layer out
    _Float16* Ah    = (_Float16*)carve((size_t)N_NODES * 96 * 2);  // f16 gemm input (padded)
    float*    asn   = (float*)   carve((size_t)N_NODES * 4);
    float*    adn   = (float*)   carve((size_t)N_NODES * 4);
    unsigned* emax  = (unsigned*)carve((size_t)N_NODES * 4);
    float*    denom = (float*)   carve((size_t)N_NODES * 4);
    float*    pbuf  = (float*)   carve((size_t)E_TOT * 4);
    _Float16* Wt0   = (_Float16*)carve((size_t)64 * 96 * 2);
    _Float16* Wtc   = (_Float16*)carve((size_t)3 * 64 * 64 * 2);
    float*    gsum  = (float*)   carve((size_t)N_GRAPHS * 64 * 4);
    float*    gcnt  = (float*)   carve((size_t)N_GRAPHS * 4);
    float*    t1    = (float*)   carve((size_t)N_GRAPHS * 64 * 4);
    float*    t2    = (float*)   carve((size_t)N_GRAPHS * 32 * 4);

    const int T = 256;
    const int bN    = (N_NODES + T - 1) / T;
    const int bN64  = (N_NODES * 64 + T - 1) / T;
    const int bN96  = (int)(((long long)N_NODES * 96 + T - 1) / T);
    const int bE    = (int)(((long long)E_TOT + T - 1) / T);
    const int bE32  = (int)(((long long)E_TOT * 32 + T - 1) / T);

    // stage weights (f16, transposed) and x (f16, padded 79->96)
    k_conv_weights<<<(64 * 96 + 3 * 64 * 64 + T - 1) / T, T, 0, stream>>>(W0, Wc, Wt0, Wtc);
    k_cvt_x<<<bN96, T, 0, stream>>>(x, Ah);

    // ---- 4 GAT layers ----
    for (int layer = 0; layer < 4; ++layer) {
        int Kp;
        const _Float16* Wl;
        const float *as_p, *ad_p, *b_p;
        if (layer == 0) {
            Kp = 96; Wl = Wt0; as_p = a0s; ad_p = a0d; b_p = b0;
        } else {
            Kp = 64; Wl = Wtc + (size_t)(layer - 1) * 64 * 64;
            as_p = acs + (size_t)(layer - 1) * 64;
            ad_p = acd + (size_t)(layer - 1) * 64;
            b_p  = bc  + (size_t)(layer - 1) * 64;
        }
        // h = in @ W  (WMMA)
        k_gemm_wmma<<<N_NODES / 32, 128, 0, stream>>>(Ah, Wl, hA, Kp);
        // per-node attention dot products
        k_node_dots<<<bN, T, 0, stream>>>(hA, as_p, ad_p, asn, adn);
        // segment softmax + aggregation over edges
        k_zero_layer<<<bN64, T, 0, stream>>>(hB, emax, denom);
        k_edge_score<<<bE, T, 0, stream>>>(ei, asn, adn, pbuf, emax);
        k_edge_exp  <<<bE, T, 0, stream>>>(ei, pbuf, emax, denom);
        k_edge_aggr <<<bE32, T, 0, stream>>>(ei, pbuf, denom, hA, hB);
        // bias + relu (+ f16 copy for next layer's GEMM, row stride 64)
        k_finalize<<<bN64, T, 0, stream>>>(hB, b_p, Ah);
    }

    // ---- global mean pool ----
    k_zero_pool<<<(N_GRAPHS * 64 + T - 1) / T, T, 0, stream>>>(gsum, gcnt);
    k_pool_acc<<<bN64, T, 0, stream>>>(hB, bat, gsum, gcnt);
    k_pool_div<<<(N_GRAPHS * 64 + T - 1) / T, T, 0, stream>>>(gsum, gcnt);

    // ---- MLP head + sigmoid ----
    k_mlp<<<(N_GRAPHS * 64 + T - 1) / T, T, 0, stream>>>(gsum, l1w, l1b, t1, 64, 64);
    k_mlp<<<(N_GRAPHS * 32 + T - 1) / T, T, 0, stream>>>(t1, l2w, l2b, t2, 64, 32);
    k_mlp3_sigmoid<<<(N_GRAPHS + T - 1) / T, T, 0, stream>>>(t2, l3w, l3b, (float*)d_out);
}
